// HtrgGraphAttentionLayer_1279900254939
// MI455X (gfx1250) — matrix operand order repaired and checked by reading, not compiled, not run on hardware
//
#include <hip/hip_runtime.h>
#include <hip/hip_bf16.h>

typedef __attribute__((ext_vector_type(16))) _Float16 v16h;
typedef __attribute__((ext_vector_type(8)))  float    v8f;

#define Bsz 4
#define Nn  512
#define N1h 256
#define Dd  64
#define Oo  64

// Branch-free transcendentals on the hardware TRANS pipe (v_exp_f32 / v_rcp_f32).
__device__ __forceinline__ float fast_exp(float x) {
  return __builtin_amdgcn_exp2f(x * 1.4426950408889634f);
}
__device__ __forceinline__ float fast_tanh(float x) {
  float xc = fminf(fmaxf(x, -9.0f), 9.0f);
  float e = __builtin_amdgcn_exp2f(xc * 2.885390081777927f);   // e^(2x)
  return (e - 1.0f) * __builtin_amdgcn_rcpf(e + 1.0f);
}

// ---------------------------------------------------------------------------
// Kernel 1: x = concat(x1@Wt1+b1, x2@Wt2+b2)  -> fp32 + f16 copies in ws
// ---------------------------------------------------------------------------
__global__ void prep_x_kernel(const float* __restrict__ x1, const float* __restrict__ x2,
                              const float* __restrict__ Wt1, const float* __restrict__ bt1,
                              const float* __restrict__ Wt2, const float* __restrict__ bt2,
                              float* __restrict__ xf, _Float16* __restrict__ xh) {
  int idx = blockIdx.x * 256 + threadIdx.x;      // < 131072
  int b = idx / (Nn * Dd);
  int r = (idx / Dd) % Nn;
  int o = idx % Dd;
  const float* src;
  const float* W;
  const float* bias;
  if (r < N1h) { src = x1 + (b * N1h + r) * Dd;         W = Wt1; bias = bt1; }
  else         { src = x2 + (b * N1h + (r - N1h)) * Dd; W = Wt2; bias = bt2; }
  float acc = bias[o];
  #pragma unroll 8
  for (int d = 0; d < Dd; ++d) acc += src[d] * W[d * Oo + o];
  xf[idx] = acc;
  xh[idx] = (_Float16)acc;
}

// W_att (k-major [k][o]) -> f16 transposed [o][k] so B-fragments are contiguous
__global__ void prep_watt_kernel(const float* __restrict__ Watt, _Float16* __restrict__ WT) {
  int idx = blockIdx.x * 256 + threadIdx.x;      // < 4096
  int o = idx / Dd, k = idx % Dd;
  WT[o * Dd + k] = (_Float16)Watt[k * Oo + o];
}

// ---------------------------------------------------------------------------
// Kernel 2: master branch (per-batch, tiny). Writes master_new to out+131072.
// ---------------------------------------------------------------------------
__global__ void master_kernel(const float* __restrict__ xf,
                              const float* __restrict__ WattM, const float* __restrict__ battM,
                              const float* __restrict__ wM,
                              const float* __restrict__ WpaM, const float* __restrict__ bpaM,
                              const float* __restrict__ WpoM, const float* __restrict__ bpoM,
                              float* __restrict__ out) {
  __shared__ float Wm[Dd * Oo];
  __shared__ float ms[Dd];
  __shared__ float zs[Nn];
  __shared__ float aM[Nn];
  __shared__ float red[256];
  __shared__ float parts[4][Dd];
  __shared__ float aggm[Dd];
  __shared__ float bam[Oo], wms[Oo];

  const int b = blockIdx.x, t = threadIdx.x;
  const float* xb = xf + b * Nn * Dd;

  for (int k = t; k < Dd * Oo; k += 256) Wm[k] = WattM[k];
  if (t < Oo) { bam[t] = battM[t]; wms[t] = wM[t]; }
  // master = mean over n
  {
    int g = t >> 6, d = t & 63;
    float p = 0.f;
    for (int n = g * 128; n < g * 128 + 128; ++n) p += xb[n * Dd + d];
    parts[g][d] = p;
  }
  __syncthreads();
  if (t < Dd) ms[t] = (parts[0][t] + parts[1][t] + parts[2][t] + parts[3][t]) * (1.0f / 512.0f);
  __syncthreads();
  // z[n] = sum_o tanh((x_n*master)@WattM + b)[o] * wM[o]
  for (int n = t; n < Nn; n += 256) {
    const float* xr = xb + n * Dd;
    float z = 0.f;
    for (int o = 0; o < Oo; ++o) {
      float acc = bam[o];
      #pragma unroll 8
      for (int d = 0; d < Dd; ++d) acc += xr[d] * ms[d] * Wm[d * Oo + o];
      z += fast_tanh(acc) * wms[o];
    }
    zs[n] = z;
  }
  __syncthreads();
  // softmax over n
  float m1 = -3.4e38f;
  for (int n = t; n < Nn; n += 256) m1 = fmaxf(m1, zs[n]);
  red[t] = m1; __syncthreads();
  for (int s = 128; s > 0; s >>= 1) { if (t < s) red[t] = fmaxf(red[t], red[t + s]); __syncthreads(); }
  float mx = red[0];
  __syncthreads();
  float s1 = 0.f;
  for (int n = t; n < Nn; n += 256) { float e = fast_exp(zs[n] - mx); aM[n] = e; s1 += e; }
  red[t] = s1; __syncthreads();
  for (int s = 128; s > 0; s >>= 1) { if (t < s) red[t] += red[t + s]; __syncthreads(); }
  float inv = 1.0f / red[0];
  // agg_m = sum_n aM[n]*x[n]
  {
    int g = t >> 6, d = t & 63;
    float p = 0.f;
    for (int n = g * 128; n < g * 128 + 128; ++n) p += aM[n] * xb[n * Dd + d];
    parts[g][d] = p;
  }
  __syncthreads();
  if (t < Dd) aggm[t] = (parts[0][t] + parts[1][t] + parts[2][t] + parts[3][t]) * inv;
  __syncthreads();
  if (t < Oo) {
    int o = t;
    float acc = bpaM[o] + bpoM[o];
    #pragma unroll 8
    for (int d = 0; d < Dd; ++d) acc += aggm[d] * WpaM[d * Oo + o] + ms[d] * WpoM[d * Oo + o];
    out[2 * Bsz * N1h * Oo + b * Oo + o] = acc;   // offset 131072
  }
}

// ---------------------------------------------------------------------------
// Kernel 3: fused attention row. One workgroup per (b,i); 8 waves.
// scores(512) via WMMA on pair-products, softmax, agg = att@x, output proj.
// ---------------------------------------------------------------------------
__global__ void __launch_bounds__(256)
attn_kernel(const float* __restrict__ xf, const _Float16* __restrict__ xh,
            const _Float16* __restrict__ WT, const float* __restrict__ batt,
            const float* __restrict__ w11, const float* __restrict__ w22,
            const float* __restrict__ w12,
            const float* __restrict__ Wpa, const float* __restrict__ bpa,
            const float* __restrict__ Wpo, const float* __restrict__ bpo,
            const float* __restrict__ gamma, const float* __restrict__ beta,
            float* __restrict__ out) {
  __shared__ __attribute__((aligned(16))) _Float16 xsh[Nn * Dd];   // 64 KB
  __shared__ float scores[Nn];
  __shared__ float att[Nn];
  __shared__ float red[256];
  __shared__ float parts[4][Dd];
  __shared__ float aggs[Dd];
  __shared__ float wsel3[3 * Oo];   // w11 | w12 | w22
  __shared__ float battS[Oo];

  const int t = threadIdx.x;
  const int b = blockIdx.x / Nn;
  const int i = blockIdx.x % Nn;

  // stage batch's x (f16) into LDS: 4096 int4s
  {
    const int4* src = (const int4*)(xh + b * Nn * Dd);
    int4* dst = (int4*)xsh;
    for (int k = t; k < (Nn * Dd) / 8; k += 256) dst[k] = src[k];
  }
  if (t < Oo) {
    wsel3[t] = w11[t]; wsel3[Oo + t] = w12[t]; wsel3[2 * Oo + t] = w22[t];
    battS[t] = batt[t];
  }
  __syncthreads();

  const int wvid = t >> 5;
  const int lane = t & 31;
  const int hl = lane >> 4;        // 16-lane half
  const int mrow = lane & 15;      // row within tile (A) / column o (B,C)
  const int olane = lane & 15;

  // B fragments (W_att^T), constant per workgroup: lane holds 16 contiguous k
  v16h bfrag[4][2];
  #pragma unroll
  for (int ot = 0; ot < 4; ++ot)
    #pragma unroll
    for (int ks = 0; ks < 2; ++ks)
      bfrag[ot][ks] = *(const v16h*)(WT + (ot * 16 + olane) * Dd + ks * 32 + hl * 16);

  const _Float16* xi = xsh + i * Dd;
  const int iside = (i < N1h) ? 0 : 1;

  for (int jt = wvid; jt < Nn / 16; jt += 8) {
    const _Float16* xj = xsh + (jt * 16 + mrow) * Dd;
    // A fragments: pair product x_i[k]*x_j[k], lane holds 16 contiguous k
    v16h afr[2];
    #pragma unroll
    for (int ks = 0; ks < 2; ++ks) {
      const int kb = ks * 32 + hl * 16;
      #pragma unroll
      for (int p = 0; p < 8; ++p) {
        const int kp = kb + 2 * p;
        afr[ks][2 * p]     = (_Float16)((float)xi[kp]     * (float)xj[kp]);
        afr[ks][2 * p + 1] = (_Float16)((float)xi[kp + 1] * (float)xj[kp + 1]);
      }
    }
    const int jside = (jt < N1h / 16) ? 0 : 1;
    const float* wsel = (iside == 0) ? (jside == 0 ? wsel3 : wsel3 + Oo)
                                     : (jside == 0 ? wsel3 + Oo : wsel3 + 2 * Oo);
    float rowacc[8] = {0.f, 0.f, 0.f, 0.f, 0.f, 0.f, 0.f, 0.f};
    #pragma unroll
    for (int ot = 0; ot < 4; ++ot) {
      v8f c = {};
      c = __builtin_amdgcn_wmma_f32_16x16x32_f16(false, afr[0], false, bfrag[ot][0],
                                                 (short)0, c, false, false);
      c = __builtin_amdgcn_wmma_f32_16x16x32_f16(false, afr[1], false, bfrag[ot][1],
                                                 (short)0, c, false, false);
      const float wv = wsel[ot * 16 + olane];
      const float bb = battS[ot * 16 + olane];
      #pragma unroll
      for (int r = 0; r < 8; ++r) rowacc[r] += wv * fast_tanh(c[r] + bb);
    }
    // reduce over the 16 lanes (o-dimension) of each half
    #pragma unroll
    for (int r = 0; r < 8; ++r) {
      float v = rowacc[r];
      v += __shfl_xor(v, 1, 32);
      v += __shfl_xor(v, 2, 32);
      v += __shfl_xor(v, 4, 32);
      v += __shfl_xor(v, 8, 32);
      if (olane == 0) scores[jt * 16 + hl * 8 + r] = v;   // lanes 0 & 16 write
    }
  }
  __syncthreads();

  // softmax over j (TEMP == 1)
  float m1 = -3.4e38f;
  for (int j = t; j < Nn; j += 256) m1 = fmaxf(m1, scores[j]);
  red[t] = m1; __syncthreads();
  for (int s = 128; s > 0; s >>= 1) { if (t < s) red[t] = fmaxf(red[t], red[t + s]); __syncthreads(); }
  float mx = red[0];
  __syncthreads();
  float s1 = 0.f;
  for (int j = t; j < Nn; j += 256) { float e = fast_exp(scores[j] - mx); att[j] = e; s1 += e; }
  red[t] = s1; __syncthreads();
  for (int s = 128; s > 0; s >>= 1) { if (t < s) red[t] += red[t + s]; __syncthreads(); }
  const float inv = 1.0f / red[0];

  // agg[d] = sum_j att[j] * x[j,d]
  {
    int g = t >> 6, d = t & 63;
    float p = 0.f;
    for (int j = g * 128; j < g * 128 + 128; ++j) p += att[j] * (float)xsh[j * Dd + d];
    parts[g][d] = p;
  }
  __syncthreads();
  if (t < Dd) aggs[t] = (parts[0][t] + parts[1][t] + parts[2][t] + parts[3][t]) * inv;
  __syncthreads();

  // y[i,o] = selu( (agg@Wpa + b_pa + x_i@Wpo + b_po) * gamma/sqrt(1+eps) + beta )
  if (t < Oo) {
    const int o = t;
    float acc = bpa[o] + bpo[o];
    const float* xif = xf + (b * Nn + i) * Dd;
    #pragma unroll 8
    for (int d = 0; d < Dd; ++d) acc += aggs[d] * Wpa[d * Oo + o] + xif[d] * Wpo[d * Oo + o];
    const float bnscale = rsqrtf(1.0f + 1e-5f);
    acc = acc * bnscale * gamma[o] + beta[o];
    float yv = (acc > 0.f) ? acc : 1.6732632423543772f * (fast_exp(acc) - 1.0f);
    yv *= 1.0507009873554805f;
    float* dst = (i < N1h) ? (out + (b * N1h + i) * Oo)
                           : (out + Bsz * N1h * Oo + (b * N1h + (i - N1h)) * Oo);
    dst[o] = yv;
  }
}

// ---------------------------------------------------------------------------
extern "C" void kernel_launch(void* const* d_in, const int* in_sizes, int n_in,
                              void* d_out, int out_size, void* d_ws, size_t ws_size,
                              hipStream_t stream) {
  const float* x1    = (const float*)d_in[0];
  const float* x2    = (const float*)d_in[1];
  const float* Wt1   = (const float*)d_in[2];
  const float* bt1   = (const float*)d_in[3];
  const float* Wt2   = (const float*)d_in[4];
  const float* bt2   = (const float*)d_in[5];
  const float* Watt  = (const float*)d_in[6];
  const float* batt  = (const float*)d_in[7];
  const float* WattM = (const float*)d_in[8];
  const float* battM = (const float*)d_in[9];
  const float* w11   = (const float*)d_in[10];
  const float* w22   = (const float*)d_in[11];
  const float* w12   = (const float*)d_in[12];
  const float* wM    = (const float*)d_in[13];
  const float* Wpa   = (const float*)d_in[14];
  const float* bpa   = (const float*)d_in[15];
  const float* Wpo   = (const float*)d_in[16];
  const float* bpo   = (const float*)d_in[17];
  const float* WpaM  = (const float*)d_in[18];
  const float* bpaM  = (const float*)d_in[19];
  const float* WpoM  = (const float*)d_in[20];
  const float* bpoM  = (const float*)d_in[21];
  const float* gamma = (const float*)d_in[22];
  const float* beta  = (const float*)d_in[23];

  char* ws = (char*)d_ws;
  float*    xf = (float*)ws;                                  // 524288 B
  _Float16* xh = (_Float16*)(ws + 524288);                    // 262144 B
  _Float16* WT = (_Float16*)(ws + 524288 + 262144);           // 8192 B

  float* out = (float*)d_out;

  prep_x_kernel<<<dim3((Bsz * Nn * Dd) / 256), dim3(256), 0, stream>>>(
      x1, x2, Wt1, bt1, Wt2, bt2, xf, xh);
  prep_watt_kernel<<<dim3((Dd * Oo) / 256), dim3(256), 0, stream>>>(Watt, WT);
  master_kernel<<<dim3(Bsz), dim3(256), 0, stream>>>(
      xf, WattM, battM, wM, WpaM, bpaM, WpoM, bpoM, out);
  attn_kernel<<<dim3(Bsz * Nn), dim3(256), 0, stream>>>(
      xf, xh, WT, batt, w11, w22, w12, Wpa, bpa, Wpo, bpo, gamma, beta, out);
}
